// TransConvLayer_78176994722522
// MI455X (gfx1250) — matrix operand order repaired and checked by reading, not compiled, not run on hardware
//
#include <hip/hip_runtime.h>
#include <hip/hip_bf16.h>

// ---------------------------------------------------------------------------
// Problem constants
// ---------------------------------------------------------------------------
#define N_NODES 32768
#define IN_CH   1024
#define OC      1024          // OUT_CH * HEADS
#define HEADS   8
#define HD      128           // OUT_CH per head
#define WSZ     (1024*1024)   // one weight matrix element count

// ---------------------------------------------------------------------------
// Types / helpers
// ---------------------------------------------------------------------------
typedef __attribute__((ext_vector_type(16))) __bf16 v16bf;
typedef __attribute__((ext_vector_type(8)))  float  v8f;

static __device__ __forceinline__ unsigned short f2bf(float f) {
    union { __bf16 h; unsigned short u; } c;
    c.h = (__bf16)f;
    return c.u;
}
static __device__ __forceinline__ float bf2f(unsigned short u) {
    union { __bf16 h; unsigned short u; } c;
    c.u = u;
    return (float)c.h;
}
static __device__ __forceinline__ unsigned pack2(unsigned short lo, unsigned short hi) {
    return (unsigned)lo | ((unsigned)hi << 16);
}

// D = A(16x32 bf16) * B(32x16 bf16) + C(16x16 f32)
static __device__ __forceinline__ v8f wmma_bf16(v16bf a, v16bf b, v8f c) {
    return __builtin_amdgcn_wmma_f32_16x16x32_bf16(
        /*neg_a=*/false, a, /*neg_b=*/false, b,
        /*c_mod=*/(short)0, c, /*reuse_a=*/false, /*reuse_b=*/false);
}

static __device__ __forceinline__ void atomAddF(float* p, float v) {
    unsafeAtomicAdd(p, v);   // native global_atomic_add_f32
}

// ---------------------------------------------------------------------------
// Fragment swizzle (contiguous k order used by this code: k = 16*(lane>>4) + j, wave32)
//
// A matrix 16x32 bf16 (per 16-row group):
//   lane = (m&15) + 16*(k>=16)    ; half j = k&15
//   -> 8 consecutive k (k%8==0) are contiguous halves of ONE lane (16B chunk)
// B matrix 32x16 bf16 (per 16-col group):
//   lane = (n&15) + 16*(k>=16)    ; half j = k&15
//   -> same contiguous-8-k-per-lane property
// Each lane's 16 halves are contiguous (32B) -> 2x b128 load per fragment.
// ---------------------------------------------------------------------------
static __device__ __forceinline__ int a_idx(int mm, int kk) {
    int mgrp = mm >> 4;
    int lane = (mm & 15) + (((kk >> 4) & 1) << 4);
    int j    = kk & 15;
    return (((mgrp << 5) + lane) << 4) + j;
}
static __device__ __forceinline__ int b_idx(int kk, int nn) {
    int ngrp = nn >> 4;
    int lane = (nn & 15) + (((kk >> 4) & 1) << 4);
    int j    = kk & 15;
    return (((ngrp << 5) + lane) << 4) + j;
}
static __device__ __forceinline__ v16bf fragLDS(const unsigned short* s, int grp, int lane) {
    return *reinterpret_cast<const v16bf*>(s + (((grp << 5) + lane) << 4));
}
// B fragments pre-swizzled in GLOBAL memory: [(slot)*32 + lane]*16 + j
static __device__ __forceinline__ v16bf fragGLB(const unsigned short* g, int slot, int lane) {
    return *reinterpret_cast<const v16bf*>(g + (((size_t)(slot << 5) + lane) << 4));
}

// ---------------------------------------------------------------------------
// Kernel 1: weight split fp32 -> bf16(hi)+bf16(lo), written directly in the
// WMMA-B fragment layout; also zero the atomic accumulators.
//   Swizzled layout per matrix: [(kt*64 + ngrp)*32 + lane]*16 + j
// ---------------------------------------------------------------------------
__global__ __launch_bounds__(256)
void prep_kernel(const float* __restrict__ Wq, const float* __restrict__ Wk,
                 const float* __restrict__ Wv,
                 unsigned short* __restrict__ Whi, unsigned short* __restrict__ Wlo,
                 float* __restrict__ zbase, int nzero) {
    int idx = blockIdx.x * 256 + threadIdx.x;
    if (idx < nzero) zbase[idx] = 0.0f;
    if (idx < 3 * WSZ) {
        int w   = idx / WSZ;
        int off = idx - w * WSZ;
        const float* src = (w == 0) ? Wq : ((w == 1) ? Wk : Wv);
        float f = src[off];
        unsigned short h = f2bf(f);
        unsigned short l = f2bf(f - bf2f(h));
        int k = off >> 10, n = off & 1023;
        int kt = k >> 5, kk = k & 31, ng = n >> 4;
        int lane = (n & 15) + (((kk >> 4) & 1) << 4);
        int j    = kk & 15;
        size_t sidx = (size_t)w * WSZ + ((((kt << 6) + ng) << 5) + lane) * 16 + j;
        Whi[sidx] = h;
        Wlo[sidx] = l;
    }
}

// ---------------------------------------------------------------------------
// Kernel 2: projection GEMM  [32768 x 1024] @ [1024 x 1024] + bias
//   block tile 128x128, BK=32, 8 waves of 64x32 (8 wmma accumulators each)
//   A: fp32 -> bf16 (hi[,lo]) packed ds_store_b128 into LDS fragment layout
//   B: direct global fragment loads (pre-swizzled, L2-resident) -- no LDS
//   EPI 0: Q -> bf16 row-major + sum(x^2)
//   EPI 1: K -> bf16 TRANSPOSED kT[col][row] (packed b128) + sum(x^2) + col sums
//   EPI 2: V -> f32 row-major + bf16 TRANSPOSED vT[col][row]; SPLIT (3 wmma)
// ---------------------------------------------------------------------------
template<int EPI, bool SPLIT>
__global__ __launch_bounds__(256)
void gemm_kernel(const float* __restrict__ X,
                 const unsigned short* __restrict__ Whi,
                 const unsigned short* __restrict__ Wlo,
                 const float* __restrict__ bias,
                 void* __restrict__ outp,
                 unsigned short* __restrict__ outT,
                 float* __restrict__ sumsq,
                 float* __restrict__ kssum) {
    __shared__ unsigned short Ah[4096], Al[4096];
    __shared__ float s_col[128];
    __shared__ float s_red;

    const int tid  = threadIdx.x;
    const int lane = tid & 31;
    const int wid  = tid >> 5;
    const int wm   = wid >> 2;   // 0..1
    const int wn   = wid & 3;    // 0..3
    const int m0   = (blockIdx.x >> 3) * 128;
    const int n0   = (blockIdx.x & 7) * 128;

    if (tid < 128) s_col[tid] = 0.0f;
    if (tid == 0)  s_red = 0.0f;

    const v8f vzero = {0.f,0.f,0.f,0.f,0.f,0.f,0.f,0.f};
    v8f acc[4][2];
#pragma unroll
    for (int mi = 0; mi < 4; ++mi)
#pragma unroll
        for (int ni = 0; ni < 2; ++ni) acc[mi][ni] = vzero;

    for (int kt = 0; kt < IN_CH / 32; ++kt) {
        const int k0 = kt * 32;
        if (kt + 1 < IN_CH / 32)
            __builtin_prefetch(X + (size_t)(m0 + (tid >> 1)) * IN_CH + k0 + 32, 0, 1);

        // ---- A tile: 128x32 fp32 -> bf16 packed 16B LDS stores ----
#pragma unroll
        for (int c = 0; c < 2; ++c) {
            int ch = tid + c * 256;              // 0..511
            int r  = ch >> 2;                    // row 0..127
            int kb = (ch & 3) * 8;               // k-chunk base
            const float* src = X + (size_t)(m0 + r) * IN_CH + k0 + kb;
            float4 x0 = *reinterpret_cast<const float4*>(src);
            float4 x1 = *reinterpret_cast<const float4*>(src + 4);
            const float xs[8] = {x0.x,x0.y,x0.z,x0.w,x1.x,x1.y,x1.z,x1.w};
            unsigned short hs[8];
#pragma unroll
            for (int i = 0; i < 8; ++i) hs[i] = f2bf(xs[i]);
            uint4 hp = { pack2(hs[0],hs[1]), pack2(hs[2],hs[3]),
                         pack2(hs[4],hs[5]), pack2(hs[6],hs[7]) };
            *reinterpret_cast<uint4*>(&Ah[a_idx(r, kb)]) = hp;
            if (SPLIT) {
                unsigned short ls[8];
#pragma unroll
                for (int i = 0; i < 8; ++i) ls[i] = f2bf(xs[i] - bf2f(hs[i]));
                uint4 lp = { pack2(ls[0],ls[1]), pack2(ls[2],ls[3]),
                             pack2(ls[4],ls[5]), pack2(ls[6],ls[7]) };
                *reinterpret_cast<uint4*>(&Al[a_idx(r, kb)]) = lp;
            }
        }
        __syncthreads();

        // ---- B fragments straight from global (pre-swizzled) ----
        v16bf ah[4], al[4];
#pragma unroll
        for (int mi = 0; mi < 4; ++mi) {
            ah[mi] = fragLDS(Ah, wm * 4 + mi, lane);
            if (SPLIT) al[mi] = fragLDS(Al, wm * 4 + mi, lane);
        }
#pragma unroll
        for (int ni = 0; ni < 2; ++ni) {
            int slot = kt * 64 + (n0 >> 4) + wn * 2 + ni;
            v16bf bh = fragGLB(Whi, slot, lane);
            v16bf bl;
            if (SPLIT) bl = fragGLB(Wlo, slot, lane);
#pragma unroll
            for (int mi = 0; mi < 4; ++mi) {
                acc[mi][ni] = wmma_bf16(ah[mi], bh, acc[mi][ni]);
                if (SPLIT) {
                    acc[mi][ni] = wmma_bf16(ah[mi], bl, acc[mi][ni]);
                    acc[mi][ni] = wmma_bf16(al[mi], bh, acc[mi][ni]);
                }
            }
        }
        __syncthreads();
    }

    // ---- epilogue ----
    // C-fragment: lane holds 8 consecutive rows of ONE column -> a packed
    // uint4 is a contiguous 16B chunk of the TRANSPOSED output.
    float local_sq = 0.0f;
#pragma unroll
    for (int mi = 0; mi < 4; ++mi) {
#pragma unroll
        for (int ni = 0; ni < 2; ++ni) {
            int colL = (wn * 2 + ni) * 16 + (lane & 15);
            int col  = n0 + colL;
            float b  = bias[col];
            int rbase = m0 + (wm * 4 + mi) * 16 + ((lane >> 4) & 1) * 8;
            float vv[8];
#pragma unroll
            for (int j = 0; j < 8; ++j) vv[j] = acc[mi][ni][j] + b;

            if (EPI == 0) {               // Q: row-major bf16
#pragma unroll
                for (int j = 0; j < 8; ++j)
                    reinterpret_cast<unsigned short*>(outp)[(size_t)(rbase + j) * OC + col] = f2bf(vv[j]);
            }
            if (EPI == 1 || EPI == 2) {   // K/V: transposed bf16, one b128 store
                unsigned short hs[8];
#pragma unroll
                for (int j = 0; j < 8; ++j) hs[j] = f2bf(vv[j]);
                uint4 tp = { pack2(hs[0],hs[1]), pack2(hs[2],hs[3]),
                             pack2(hs[4],hs[5]), pack2(hs[6],hs[7]) };
                *reinterpret_cast<uint4*>(&outT[(size_t)col * N_NODES + rbase]) = tp;
            }
            if (EPI == 2) {               // V: also fp32 row-major
#pragma unroll
                for (int j = 0; j < 8; ++j)
                    reinterpret_cast<float*>(outp)[(size_t)(rbase + j) * OC + col] = vv[j];
            }
            if (EPI == 0 || EPI == 1) {
                float csum = 0.0f;
#pragma unroll
                for (int j = 0; j < 8; ++j) { local_sq += vv[j] * vv[j]; csum += vv[j]; }
                if (EPI == 1) atomicAdd(&s_col[colL], csum);   // ds_add_f32
            }
        }
    }
    if (EPI == 0 || EPI == 1) {
        atomicAdd(&s_red, local_sq);                            // ds_add_f32
        __syncthreads();
        if (tid == 0) atomAddF(sumsq, s_red);
        if (EPI == 1 && tid < 128) atomAddF(&kssum[n0 + tid], s_col[tid]);
    }
}

// ---------------------------------------------------------------------------
// Kernel 3: kvs[h,m,d] = sum_l kT[h*128+m][l] * vT[h*128+d][l]
//   split-K (64 chunks of 512); both operands feature-major -> pure verbatim
//   b128 global->LDS staging (no conversions, no scalar scatter);
//   fp32 atomic accumulate into L2-resident kvs
// ---------------------------------------------------------------------------
__global__ __launch_bounds__(256)
void kvs_kernel(const unsigned short* __restrict__ kT,
                const unsigned short* __restrict__ vT,
                float* __restrict__ kvs) {
    __shared__ unsigned short Ah[4096], Bh[4096];

    const int tid  = threadIdx.x;
    const int lane = tid & 31;
    const int wid  = tid >> 5;
    const int wm   = wid >> 2;
    const int wn   = wid & 3;
    const int head  = blockIdx.x >> 6;
    const int chunk = blockIdx.x & 63;
    const int l0    = chunk * 512;

    const v8f vzero = {0.f,0.f,0.f,0.f,0.f,0.f,0.f,0.f};
    v8f acc[4][2];
#pragma unroll
    for (int mi = 0; mi < 4; ++mi)
#pragma unroll
        for (int ni = 0; ni < 2; ++ni) acc[mi][ni] = vzero;

    for (int it = 0; it < 16; ++it) {
        const int lt = l0 + it * 32;
#pragma unroll
        for (int c = 0; c < 2; ++c) {
            int ch = tid + c * 256;              // 0..511
            int r  = ch >> 2;                    // 0..127: A row m / B col d
            int kb = (ch & 3) * 8;               // l-chunk base
            uint4 wa = *reinterpret_cast<const uint4*>(
                kT + (size_t)(head * HD + r) * N_NODES + lt + kb);
            *reinterpret_cast<uint4*>(&Ah[a_idx(r, kb)]) = wa;
            uint4 wb = *reinterpret_cast<const uint4*>(
                vT + (size_t)(head * HD + r) * N_NODES + lt + kb);
            *reinterpret_cast<uint4*>(&Bh[b_idx(kb, r)]) = wb;
        }
        __syncthreads();
#pragma unroll
        for (int ni = 0; ni < 2; ++ni) {
            v16bf bh = fragLDS(Bh, wn * 2 + ni, lane);
#pragma unroll
            for (int mi = 0; mi < 4; ++mi) {
                v16bf ah = fragLDS(Ah, wm * 4 + mi, lane);
                acc[mi][ni] = wmma_bf16(ah, bh, acc[mi][ni]);
            }
        }
        __syncthreads();
    }
#pragma unroll
    for (int mi = 0; mi < 4; ++mi)
#pragma unroll
        for (int ni = 0; ni < 2; ++ni) {
            int col = (wn * 2 + ni) * 16 + (lane & 15);
            int rb  = (wm * 4 + mi) * 16 + ((lane >> 4) & 1) * 8;
#pragma unroll
            for (int j = 0; j < 8; ++j)
                atomAddF(&kvs[(size_t)(head * HD + rb + j) * HD + col], acc[mi][ni][j]);
        }
}

// ---------------------------------------------------------------------------
// Kernel 3b: pack kvs fp32 -> bf16 in global WMMA-B fragment layout
//   slot(h,kt,ng) = (h*4 + kt)*8 + ng  ->  [slot*32 + lane]*16 + j
// ---------------------------------------------------------------------------
__global__ __launch_bounds__(256)
void kvspack_kernel(const float* __restrict__ kvs,
                    unsigned short* __restrict__ kvsbf) {
    int idx = blockIdx.x * 256 + threadIdx.x;   // 0 .. 8*128*128-1
    int h   = idx >> 14;
    int rem = idx & 16383;
    int m   = rem >> 7;
    int d   = rem & 127;
    int kt = m >> 5, kk = m & 31, ng = d >> 4;
    int lane = (d & 15) + (((kk >> 4) & 1) << 4);
    int j    = kk & 15;
    int sidx = (((((h * 4 + kt) * 8 + ng) << 5) + lane) << 4) + j;
    kvsbf[sidx] = f2bf(kvs[idx]);
}

// ---------------------------------------------------------------------------
// Kernel 4: den[n,h] = inv_qk * (q[n,h,:] . ks_sum[h,:]) + N
// ---------------------------------------------------------------------------
__global__ __launch_bounds__(256)
void normalizer_kernel(const unsigned short* __restrict__ qbf,
                       const float* __restrict__ kssum,
                       const float* __restrict__ scal,
                       float* __restrict__ den) {
    int id = blockIdx.x * 256 + threadIdx.x;    // 0 .. 32768*8-1
    int n = id >> 3, h = id & 7;
    float inv_qk = rsqrtf(scal[0]) * rsqrtf(scal[1]);
    size_t base = (size_t)n * OC + h * HD;
    float t = 0.0f;
#pragma unroll
    for (int c = 0; c < 16; ++c) {
        uint4 w = *reinterpret_cast<const uint4*>(qbf + base + c * 8);
        const unsigned wc[4] = {w.x, w.y, w.z, w.w};
#pragma unroll
        for (int i = 0; i < 8; ++i) {
            unsigned short u = (unsigned short)((wc[i >> 1] >> ((i & 1) * 16)) & 0xFFFF);
            t += bf2f(u) * kssum[h * HD + c * 8 + i];
        }
    }
    den[id] = t * inv_qk + (float)N_NODES;
}

// ---------------------------------------------------------------------------
// Kernel 5: out[n,d] = (1/H) * sum_h (inv_qk * q[n,h,:].kvs[h,:,d] + N*v[n,h,d]) / den[n,h]
//   one block per 128-row tile; loops heads; q@kvs via WMMA (K=128)
//   A: bf16 q rows -> verbatim ds_store_b128 into LDS fragment layout
//   B: direct global fragment loads from pre-packed kvsbf (256 KB, L2)
// ---------------------------------------------------------------------------
__global__ __launch_bounds__(256)
void attn_out_kernel(const unsigned short* __restrict__ qbf,
                     const unsigned short* __restrict__ kvsbf,
                     const float* __restrict__ vf,
                     const float* __restrict__ den,
                     const float* __restrict__ scal,
                     float* __restrict__ out) {
    __shared__ unsigned short Ah[4096];

    const int tid  = threadIdx.x;
    const int lane = tid & 31;
    const int wid  = tid >> 5;
    const int wm   = wid >> 2;
    const int wn   = wid & 3;
    const int n0   = blockIdx.x * 128;
    const float inv_qk = rsqrtf(scal[0]) * rsqrtf(scal[1]);

    const v8f vzero = {0.f,0.f,0.f,0.f,0.f,0.f,0.f,0.f};
    v8f facc[4][2];
#pragma unroll
    for (int mi = 0; mi < 4; ++mi)
#pragma unroll
        for (int ni = 0; ni < 2; ++ni) facc[mi][ni] = vzero;

    for (int h = 0; h < HEADS; ++h) {
        v8f hacc[4][2];
#pragma unroll
        for (int mi = 0; mi < 4; ++mi)
#pragma unroll
            for (int ni = 0; ni < 2; ++ni) hacc[mi][ni] = vzero;

        for (int kt = 0; kt < 4; ++kt) {
            // A tile: q rows n0..n0+127, cols h*128 + kt*32 .. (verbatim 16B copies)
#pragma unroll
            for (int c = 0; c < 2; ++c) {
                int ch = tid + c * 256;          // 0..511
                int mm = ch >> 2;
                int kb = (ch & 3) * 8;
                uint4 w = *reinterpret_cast<const uint4*>(
                    qbf + (size_t)(n0 + mm) * OC + h * HD + kt * 32 + kb);
                *reinterpret_cast<uint4*>(&Ah[a_idx(mm, kb)]) = w;
            }
            __syncthreads();
#pragma unroll
            for (int ni = 0; ni < 2; ++ni) {
                int slot = (h * 4 + kt) * 8 + wn * 2 + ni;
                v16bf bh = fragGLB(kvsbf, slot, lane);
#pragma unroll
                for (int mi = 0; mi < 4; ++mi) {
                    v16bf ah = fragLDS(Ah, wm * 4 + mi, lane);
                    hacc[mi][ni] = wmma_bf16(ah, bh, hacc[mi][ni]);
                }
            }
            __syncthreads();
        }
        // combine this head
#pragma unroll
        for (int mi = 0; mi < 4; ++mi) {
#pragma unroll
            for (int ni = 0; ni < 2; ++ni) {
                int colL = (wn * 2 + ni) * 16 + (lane & 15);
                int rb   = (wm * 4 + mi) * 16 + ((lane >> 4) & 1) * 8;
#pragma unroll
                for (int j = 0; j < 8; ++j) {
                    int row = n0 + rb + j;
                    float num = hacc[mi][ni][j] * inv_qk
                              + (float)N_NODES * vf[(size_t)row * OC + h * HD + colL];
                    facc[mi][ni][j] = facc[mi][ni][j] + num / den[row * HEADS + h];
                }
            }
        }
    }
    // write output (mean over heads)
#pragma unroll
    for (int mi = 0; mi < 4; ++mi)
#pragma unroll
        for (int ni = 0; ni < 2; ++ni) {
            int colL = (wn * 2 + ni) * 16 + (lane & 15);
            int rb   = (wm * 4 + mi) * 16 + ((lane >> 4) & 1) * 8;
#pragma unroll
            for (int j = 0; j < 8; ++j)
                out[(size_t)(n0 + rb + j) * HD + colL] = facc[mi][ni][j] * (1.0f / HEADS);
        }
}

// ---------------------------------------------------------------------------
// Launch
// ---------------------------------------------------------------------------
extern "C" void kernel_launch(void* const* d_in, const int* in_sizes, int n_in,
                              void* d_out, int out_size, void* d_ws, size_t ws_size,
                              hipStream_t stream) {
    const float* query  = (const float*)d_in[0];
    const float* source = (const float*)d_in[1];
    const float* Wq_w = (const float*)d_in[2];
    const float* Wq_b = (const float*)d_in[3];
    const float* Wk_w = (const float*)d_in[4];
    const float* Wk_b = (const float*)d_in[5];
    const float* Wv_w = (const float*)d_in[6];
    const float* Wv_b = (const float*)d_in[7];

    unsigned char* ws = (unsigned char*)d_ws;
    // workspace layout (all offsets multiples of 256B)
    size_t o = 0;
    float* scal  = (float*)(ws + o); o += 256;                         // [0]=sumsq_q [1]=sumsq_k
    float* kssum = (float*)(ws + o); o += (size_t)OC * 4;              // 4 KB
    float* kvs   = (float*)(ws + o); o += (size_t)HEADS * HD * HD * 4; // 512 KB
    size_t zbytes = o;                                                 // zeroed each launch
    float* den   = (float*)(ws + o); o += (size_t)N_NODES * HEADS * 4;
    unsigned short* kvsbf = (unsigned short*)(ws + o); o += (size_t)HEADS * HD * HD * 2;
    unsigned short* Whi = (unsigned short*)(ws + o); o += (size_t)3 * WSZ * 2;
    unsigned short* Wlo = (unsigned short*)(ws + o); o += (size_t)3 * WSZ * 2;
    unsigned short* qbf = (unsigned short*)(ws + o); o += (size_t)N_NODES * OC * 2;  // row-major
    unsigned short* kT  = (unsigned short*)(ws + o); o += (size_t)N_NODES * OC * 2;  // [col][row]
    unsigned short* vT  = (unsigned short*)(ws + o); o += (size_t)N_NODES * OC * 2;  // [col][row]
    float* vf    = (float*)(ws + o); o += (size_t)N_NODES * OC * 4;                  // row-major
    (void)o; (void)ws_size; (void)in_sizes; (void)n_in; (void)out_size;

    const int nzero = (int)(zbytes / 4);

    prep_kernel<<<(3 * WSZ + 255) / 256, 256, 0, stream>>>(
        Wq_w, Wk_w, Wv_w, Whi, Wlo, scal, nzero);

    const int gemm_blocks = (N_NODES / 128) * (OC / 128);   // 2048
    gemm_kernel<0, false><<<gemm_blocks, 256, 0, stream>>>(
        query,  Whi + 0 * (size_t)WSZ, Wlo + 0 * (size_t)WSZ, Wq_b,
        qbf, nullptr, &scal[0], nullptr);
    gemm_kernel<1, false><<<gemm_blocks, 256, 0, stream>>>(
        source, Whi + 1 * (size_t)WSZ, Wlo + 1 * (size_t)WSZ, Wk_b,
        nullptr, kT, &scal[1], kssum);
    gemm_kernel<2, true><<<gemm_blocks, 256, 0, stream>>>(
        source, Whi + 2 * (size_t)WSZ, Wlo + 2 * (size_t)WSZ, Wv_b,
        vf, vT, nullptr, nullptr);

    kvs_kernel<<<HEADS * 64, 256, 0, stream>>>(kT, vT, kvs);
    kvspack_kernel<<<(HEADS * HD * HD) / 256, 256, 0, stream>>>(kvs, kvsbf);
    normalizer_kernel<<<(N_NODES * HEADS) / 256, 256, 0, stream>>>(qbf, kssum, scal, den);
    attn_out_kernel<<<N_NODES / 128, 256, 0, stream>>>(qbf, kvsbf, vf, den, scal, (float*)d_out);
}
